// HumanVAttention_53850299957483
// MI455X (gfx1250) — matrix operand order repaired and checked by reading, not compiled, not run on hardware
//
#include <hip/hip_runtime.h>

typedef __attribute__((ext_vector_type(16))) __bf16 v16bf;
typedef __attribute__((ext_vector_type(8)))  float  v8f;

#define H_   16
#define HKV_ 4
#define D_   128
#define ROT_ 64
#define T_   4096
#define B_   2
#define HID_ 2048
#define MROWS (B_*T_)          // 8192
#define SCALE_ 0.08838834764831845f
#define NEG_ -1000000000.0f

union Frag { v16bf v; unsigned u[8]; };

__device__ __forceinline__ v8f wmma_bf16(const Frag& a, const Frag& b, const v8f& c) {
  return __builtin_amdgcn_wmma_f32_16x16x32_bf16(false, a.v, false, b.v, (short)0, c, false, false);
}

// per-lane async global->LDS 16-byte copy (ASYNCcnt-tracked, no VGPR data)
__device__ __forceinline__ void async_b128(void* lds, const void* g) {
  unsigned l = (unsigned)(unsigned long long)lds;   // low 32 bits of generic ptr = LDS addr
  asm volatile("global_load_async_to_lds_b128 %0, %1, off" :: "v"(l), "v"(g) : "memory");
}
__device__ __forceinline__ void wait_async0() {
  asm volatile("s_wait_asynccnt 0x0" ::: "memory");
}

// ---------------- fp32 -> bf16 convert ----------------
__global__ __launch_bounds__(256) void cvt_f32_bf16(const float* __restrict__ src,
                                                    __bf16* __restrict__ dst, int n) {
  int i = blockIdx.x * 1024 + threadIdx.x * 4;
  if (i + 3 < n) {
    float4 f = *(const float4*)(src + i);
    dst[i+0] = (__bf16)f.x; dst[i+1] = (__bf16)f.y;
    dst[i+2] = (__bf16)f.z; dst[i+3] = (__bf16)f.w;
  }
}

// ---------------- generic GEMM: C[M,N] = A[M,K] * W[N,K]^T ----------------
// 128(M) x 64(N) workgroup tile, 8 waves of 32, each wave computes 32x32.
template<bool F32OUT>
__global__ __launch_bounds__(256) void gemm_wmma(const __bf16* __restrict__ A,
                                                 const __bf16* __restrict__ W,
                                                 void* __restrict__ Cout,
                                                 int M, int N, int K, int ntiles) {
  __shared__ __bf16 As[2][128 * 32];
  __shared__ __bf16 Bs[2][64 * 32];

  const int mt = blockIdx.x / ntiles, nt = blockIdx.x % ntiles;
  const int mbase = mt * 128, nbase = nt * 64;
  const int tid = threadIdx.x;
  const int lane = tid & 31, wave = tid >> 5;
  const int wm = wave >> 1, wn = wave & 1;       // 4x2 wave grid
  const int ln = lane & 15, lh = lane >> 4;

  v8f c[2][2];
  for (int a = 0; a < 2; ++a)
    for (int b = 0; b < 2; ++b)
      c[a][b] = (v8f){0.f,0.f,0.f,0.f,0.f,0.f,0.f,0.f};

  const int nk = K >> 5;

  // async-stage K-slab ks into buffer buf (3 x b128 per thread)
  auto stage = [&](int ks, int buf) {
    const int k0 = ks * 32;
    for (int s = 0; s < 2; ++s) {
      int chunk = tid + 256 * s;                 // 512 uint4 = 128x32 bf16
      int row = chunk >> 2, c16 = chunk & 3;
      async_b128(&As[buf][row*32 + c16*8],
                 &A[(size_t)(mbase+row)*K + k0 + c16*8]);
    }
    { int row = tid >> 2, c16 = tid & 3;         // 256 uint4 = 64x32 bf16
      async_b128(&Bs[buf][row*32 + c16*8],
                 &W[(size_t)(nbase+row)*K + k0 + c16*8]); }
  };

  stage(0, 0);

  for (int ks = 0; ks < nk; ++ks) {
    const int buf = ks & 1;
    wait_async0();              // only stage(ks)'s ops are outstanding here
    __syncthreads();
    if (ks + 1 < nk) stage(ks + 1, buf ^ 1);   // DMA overlaps the WMMAs below

    Frag a[2], bf[2];
    for (int t = 0; t < 2; ++t) {
      const int m = wm*32 + t*16 + ln;           // A: 16x32, row in lane
      for (int j = 0; j < 8; ++j) {
        int kp = (j < 4) ? (lh*8 + j*2) : (16 + lh*8 + (j-4)*2);
        a[t].u[j] = *(const unsigned*)&As[buf][m*32 + kp];
      }
      const int n = wn*32 + t*16 + ln;           // B: 32x16 (KxN), col in lane
      for (int j = 0; j < 8; ++j) {
        int kp = (j < 4) ? (lh*8 + j*2) : (16 + lh*8 + (j-4)*2);
        bf[t].u[j] = *(const unsigned*)&Bs[buf][n*32 + kp];
      }
    }
    for (int ti = 0; ti < 2; ++ti)
      for (int tj = 0; tj < 2; ++tj)
        c[ti][tj] = wmma_bf16(a[ti], bf[tj], c[ti][tj]);
  }

  for (int ti = 0; ti < 2; ++ti)
    for (int tj = 0; tj < 2; ++tj)
      for (int v = 0; v < 8; ++v) {
        int row = mbase + wm*32 + ti*16 + v + 8*lh;
        int col = nbase + wn*32 + tj*16 + ln;
        float val = c[ti][tj][v];
        if (F32OUT) ((float*)Cout)[(size_t)row*N + col] = val;
        else        ((__bf16*)Cout)[(size_t)row*N + col] = (__bf16)val;
      }
}

// ---------------- RoPE on bf16 Q/K ----------------
__global__ __launch_bounds__(256) void rope_kernel(__bf16* __restrict__ X,
                                                   const float* __restrict__ cosb,
                                                   const float* __restrict__ sinb,
                                                   int nheads) {
  int idx = blockIdx.x * 256 + threadIdx.x;
  int total = MROWS * nheads * 32;
  if (idx >= total) return;
  int j = idx & 31;
  int head = (idx >> 5) % nheads;
  int row = idx / (32 * nheads);
  __bf16* p = X + (size_t)row * nheads * D_ + head * D_;
  float x1 = (float)p[j], x2 = (float)p[j + 32];
  const float* cr = cosb + (size_t)row * ROT_;
  const float* sr = sinb + (size_t)row * ROT_;
  float r1 = x1 * cr[j]      - x2 * sr[j];
  float r2 = x2 * cr[j + 32] + x1 * sr[j + 32];
  p[j] = (__bf16)r1; p[j + 32] = (__bf16)r2;
}

// ---------------- block-sparse flash attention ----------------
// grid = B*H*nblocks (2*16*64 = 2048), 128 threads = 4 waves, wave = q-tile.
__global__ __launch_bounds__(128) void attn_kernel(const __bf16* __restrict__ Q,
                                                   const __bf16* __restrict__ Kx,
                                                   const __bf16* __restrict__ Vx,
                                                   __bf16* __restrict__ O) {
  __shared__ __bf16 qs[64 * 128];          // 16 KB
  __shared__ __bf16 ksl[2][32 * 128];      // 16 KB  (keys row-major)
  __shared__ __bf16 vt[2][128 * 32];       // 16 KB  (V transposed: [d][key])
  __shared__ __bf16 ps[4][16 * 32];        // 4 KB per-wave P scratch

  const int bid = blockIdx.x;
  const int iblk = bid & 63;
  const int h    = (bid >> 6) & 15;
  const int b    = bid >> 10;
  const int kvh  = h >> 2;
  const int tid = threadIdx.x, lane = tid & 31, wave = tid >> 5;
  const int ln = lane & 15, lh = lane >> 4;

  // async-stage Q block (64 x 128): 8 x b128 per lane
  const __bf16* Qb = Q + ((size_t)(b*T_ + iblk*64)) * (H_*D_) + h*D_;
  for (int s = 0; s < 8; ++s) {
    int chunk = tid + 128 * s;               // 1024 uint4
    int row = chunk >> 4, c8 = chunk & 15;
    async_b128(&qs[row*128 + c8*8], &Qb[(size_t)row*(H_*D_) + c8*8]);
  }

  const int nblk = (iblk >= 8) ? 9 : (iblk + 1);
  const int nchunks = nblk * 2;              // 32 keys per chunk

  auto blk_of = [&](int c) {
    int slot = c >> 1;
    return (iblk >= 8) ? (slot == 0 ? 0 : iblk - 8 + slot) : slot;
  };

  auto stage = [&](int c, int buf) {
    const int half = c & 1;
    const size_t krow0 = (size_t)b*T_ + blk_of(c)*64 + half*32;
    const __bf16* Kb = Kx + krow0 * (HKV_*D_) + kvh*D_;
    const __bf16* Vb = Vx + krow0 * (HKV_*D_) + kvh*D_;
    for (int s = 0; s < 4; ++s) {
      int chunk = tid + 128 * s;             // 512 uint4 = 32x128 bf16
      int row = chunk >> 4, c8 = chunk & 15;
      async_b128(&ksl[buf][row*128 + c8*8], &Kb[(size_t)row*(HKV_*D_) + c8*8]);
      uint4 vv = *(const uint4*)&Vb[(size_t)row*(HKV_*D_) + c8*8];
      const __bf16* ve = (const __bf16*)&vv;
      for (int e = 0; e < 8; ++e) vt[buf][(c8*8 + e)*32 + row] = ve[e];
    }
  };

  float m_run[8], l_run[8];
  v8f o[8];
  for (int v = 0; v < 8; ++v) { m_run[v] = -3.0e38f; l_run[v] = 0.f; }
  for (int dt = 0; dt < 8; ++dt) o[dt] = (v8f){0.f,0.f,0.f,0.f,0.f,0.f,0.f,0.f};

  stage(0, 0);

  for (int cix = 0; cix < nchunks; ++cix) {
    const int buf = cix & 1;
    wait_async0();               // stage(cix) (and Q on cix==0) now resident
    __syncthreads();
    if (cix + 1 < nchunks) stage(cix + 1, buf ^ 1);   // overlaps compute

    // ---- S = Q K^T over D=128 (4 WMMA k-steps, 2 key tiles) ----
    v8f s0 = (v8f){0.f,0.f,0.f,0.f,0.f,0.f,0.f,0.f};
    v8f s1 = (v8f){0.f,0.f,0.f,0.f,0.f,0.f,0.f,0.f};
    for (int ksx = 0; ksx < 4; ++ksx) {
      Frag qa, kb0, kb1;
      const int m = wave*16 + ln;
      for (int j = 0; j < 8; ++j) {
        int kp = (j < 4) ? (lh*8 + j*2) : (16 + lh*8 + (j-4)*2);
        qa.u[j] = *(const unsigned*)&qs[m*128 + ksx*32 + kp];
      }
      for (int j = 0; j < 8; ++j) {
        int kp = (j < 4) ? (lh*8 + j*2) : (16 + lh*8 + (j-4)*2);
        kb0.u[j] = *(const unsigned*)&ksl[buf][(ln      )*128 + ksx*32 + kp];
        kb1.u[j] = *(const unsigned*)&ksl[buf][(16 + ln )*128 + ksx*32 + kp];
      }
      s0 = wmma_bf16(qa, kb0, s0);
      s1 = wmma_bf16(qa, kb1, s1);
    }

    // ---- scale + causal mask (self block only) ----
    const bool self = (blk_of(cix) == iblk);
    const int kbase = (cix & 1) * 32;
    for (int v = 0; v < 8; ++v) {
      int qrow = wave*16 + v + 8*lh;
      float a0 = s0[v] * SCALE_, a1 = s1[v] * SCALE_;
      if (self) {
        if (kbase + ln      > qrow) a0 = NEG_;
        if (kbase + 16 + ln > qrow) a1 = NEG_;
      }
      s0[v] = a0; s1[v] = a1;
    }

    // ---- online softmax (row reductions across 16-lane groups) ----
    float corr[8];
    for (int v = 0; v < 8; ++v) {
      float mx = fmaxf(s0[v], s1[v]);
      for (int off = 8; off >= 1; off >>= 1) mx = fmaxf(mx, __shfl_xor(mx, off, 16));
      float mnew = fmaxf(m_run[v], mx);
      float cf = __expf(m_run[v] - mnew);
      float p0 = __expf(s0[v] - mnew);
      float p1 = __expf(s1[v] - mnew);
      float rs = p0 + p1;
      for (int off = 8; off >= 1; off >>= 1) rs += __shfl_xor(rs, off, 16);
      l_run[v] = l_run[v] * cf + rs;
      m_run[v] = mnew;
      corr[v] = cf;
      s0[v] = p0; s1[v] = p1;
    }

    // ---- P -> A-layout via per-wave LDS scratch ----
    for (int v = 0; v < 8; ++v) {
      int m = v + 8*lh;
      ps[wave][m*32 + ln]      = (__bf16)s0[v];
      ps[wave][m*32 + 16 + ln] = (__bf16)s1[v];
    }
    asm volatile("s_wait_dscnt 0x0" ::: "memory");
    Frag pa;
    for (int j = 0; j < 8; ++j) {
      int kp = (j < 4) ? (lh*8 + j*2) : (16 + lh*8 + (j-4)*2);
      pa.u[j] = *(const unsigned*)&ps[wave][ln*32 + kp];
    }

    // ---- O = O*corr + P V (8 d-tiles) ----
    for (int dt = 0; dt < 8; ++dt) {
      for (int v = 0; v < 8; ++v) o[dt][v] *= corr[v];
      Frag vb;
      for (int j = 0; j < 8; ++j) {
        int kp = (j < 4) ? (lh*8 + j*2) : (16 + lh*8 + (j-4)*2);
        vb.u[j] = *(const unsigned*)&vt[buf][(dt*16 + ln)*32 + kp];
      }
      o[dt] = wmma_bf16(pa, vb, o[dt]);
    }
  }

  // ---- normalize + store bf16 to attention output [B*T, H*D] ----
  __bf16* Ob = O + ((size_t)(b*T_ + iblk*64)) * (H_*D_) + h*D_;
  for (int dt = 0; dt < 8; ++dt)
    for (int v = 0; v < 8; ++v) {
      int qrow = wave*16 + v + 8*lh;
      Ob[(size_t)qrow*(H_*D_) + dt*16 + ln] = (__bf16)(o[dt][v] / l_run[v]);
    }
}

// ---------------- host orchestration ----------------
extern "C" void kernel_launch(void* const* d_in, const int* in_sizes, int n_in,
                              void* d_out, int out_size, void* d_ws, size_t ws_size,
                              hipStream_t stream) {
  const float* hidden = (const float*)d_in[0];
  const float* cosb   = (const float*)d_in[1];
  const float* sinb   = (const float*)d_in[2];
  const float* Wq     = (const float*)d_in[3];
  const float* Wk     = (const float*)d_in[4];
  const float* Wv     = (const float*)d_in[5];
  const float* Wo     = (const float*)d_in[6];

  char* ws = (char*)d_ws;
  __bf16* hb  = (__bf16*)ws; ws += (size_t)MROWS * HID_ * 2;
  __bf16* wqb = (__bf16*)ws; ws += (size_t)(H_*D_)  * HID_ * 2;
  __bf16* wkb = (__bf16*)ws; ws += (size_t)(HKV_*D_)* HID_ * 2;
  __bf16* wvb = (__bf16*)ws; ws += (size_t)(HKV_*D_)* HID_ * 2;
  __bf16* wob = (__bf16*)ws; ws += (size_t)HID_ * (H_*D_) * 2;
  __bf16* qb  = (__bf16*)ws; ws += (size_t)MROWS * (H_*D_)  * 2;
  __bf16* kb  = (__bf16*)ws; ws += (size_t)MROWS * (HKV_*D_)* 2;
  __bf16* vb  = (__bf16*)ws; ws += (size_t)MROWS * (HKV_*D_)* 2;
  __bf16* aob = (__bf16*)ws; ws += (size_t)MROWS * (H_*D_)  * 2;

  // converts
  {
    int n;
    n = MROWS * HID_;        cvt_f32_bf16<<<n/1024, 256, 0, stream>>>(hidden, hb, n);
    n = H_*D_ * HID_;        cvt_f32_bf16<<<n/1024, 256, 0, stream>>>(Wq, wqb, n);
    n = HKV_*D_ * HID_;      cvt_f32_bf16<<<n/1024, 256, 0, stream>>>(Wk, wkb, n);
    n = HKV_*D_ * HID_;      cvt_f32_bf16<<<n/1024, 256, 0, stream>>>(Wv, wvb, n);
    n = HID_ * H_*D_;        cvt_f32_bf16<<<n/1024, 256, 0, stream>>>(Wo, wob, n);
  }

  // QKV projections (bf16 out)
  gemm_wmma<false><<<(MROWS/128) * (2048/64), 256, 0, stream>>>(hb, wqb, qb, MROWS, 2048, HID_, 2048/64);
  gemm_wmma<false><<<(MROWS/128) * (512/64),  256, 0, stream>>>(hb, wkb, kb, MROWS,  512, HID_,  512/64);
  gemm_wmma<false><<<(MROWS/128) * (512/64),  256, 0, stream>>>(hb, wvb, vb, MROWS,  512, HID_,  512/64);

  // RoPE
  rope_kernel<<<(MROWS*H_*32)/256,   256, 0, stream>>>(qb, cosb, sinb, H_);
  rope_kernel<<<(MROWS*HKV_*32)/256, 256, 0, stream>>>(kb, cosb, sinb, HKV_);

  // block-sparse attention
  attn_kernel<<<B_*H_*(T_/64), 128, 0, stream>>>(qb, kb, vb, aob);

  // output projection (f32 out -> d_out)
  gemm_wmma<true><<<(MROWS/128) * (2048/64), 256, 0, stream>>>(aob, wob, d_out, MROWS, 2048, HID_, 2048/64);
}
